// sLawformerMoudle_59846074302772
// MI455X (gfx1250) — matrix operand order repaired and checked by reading, not compiled, not run on hardware
//
#include <hip/hip_runtime.h>
#include <hip/hip_bf16.h>
#include <math.h>

#define D_MODEL 768
#define NHEAD   12
#define DH      64
#define FF      3072
#define NLAYER  12
#define CHUNK   256
#define LN_EPS  1e-12f

typedef __attribute__((ext_vector_type(16))) __bf16 bf16x16;
typedef __attribute__((ext_vector_type(8)))  __bf16 bf16x8;
typedef __attribute__((ext_vector_type(4)))  __bf16 bf16x4;
typedef __attribute__((ext_vector_type(8)))  float  f32x8;

union frag_t { bf16x16 v; bf16x8 h[2]; };

__device__ __forceinline__ f32x8 wmma_bf16(bf16x16 a, bf16x16 b, f32x8 c) {
  return __builtin_amdgcn_wmma_f32_16x16x32_bf16(false, a, false, b, (short)0, c, false, false);
}
__device__ __forceinline__ bf16x4 cvt4(float4 v) {
  bf16x4 r = {(__bf16)v.x, (__bf16)v.y, (__bf16)v.z, (__bf16)v.w};
  return r;
}
__device__ __forceinline__ float f4get(const float4& v, int i) {
  return i == 0 ? v.x : (i == 1 ? v.y : (i == 2 ? v.z : v.w));
}
// uniform base + 32-bit byte offset -> saddr addressing
__device__ __forceinline__ const float4* f4at(const float* p, unsigned eoff) {
  return (const float4*)((const char*)p + eoff * 4u);
}
__device__ __forceinline__ const float* fatc(const float* p, unsigned eoff) {
  return (const float*)((const char*)p + eoff * 4u);
}
__device__ __forceinline__ float* fat(float* p, unsigned eoff) {
  return (float*)((char*)p + eoff * 4u);
}
__device__ __forceinline__ const __bf16* bfat(const __bf16* p, unsigned eoff) {
  return (const __bf16*)((const char*)p + eoff * 2u);
}
// CDNA5 async global->LDS copy (ASYNCcnt-tracked), 16B per lane.
// LDS generic pointers carry the LDS byte offset in the low 32 bits.
__device__ __forceinline__ void async_ld_b128(void* lds, const void* g) {
  unsigned l = (unsigned)(size_t)lds;
  asm volatile("global_load_async_to_lds_b128 %0, %1, off"
               :: "v"(l), "v"(g) : "memory");
}
__device__ __forceinline__ void wait_async() {
  asm volatile("s_wait_asynccnt 0x0" ::: "memory");
}

// ---------------------------------------------------------------------------
// GEMM: C = act(A[M,K](bf16) @ B + bias) + residual
//   B = Bw[K,N] fp32 (weights) or Bt[N,K] bf16 (transposed activations).
// Outputs: C fp32 (optional) and/or Cbf bf16 (optional).
// act: 0=none, 1=gelu(exact), 2=*0.125.
// M%128==0, N%128==0, K%64==0. Block 128x128, KT=64, wave tile 32x64.
// ---------------------------------------------------------------------------
#define BM 128
#define BN 128
#define KT 64
#define KP 72

__global__ __launch_bounds__(256) void gemm_kernel(
    const __bf16* __restrict__ A, const float* __restrict__ Bw,
    const __bf16* __restrict__ Bt,
    const float* __restrict__ bias, const float* __restrict__ residual,
    float* __restrict__ C, __bf16* __restrict__ Cbf,
    int M, int N, int K, int act)
{
  __shared__ __bf16 As[BM][KP];
  __shared__ __bf16 Bs[BN][KP];

  const int tid = threadIdx.x;
  const int lane = tid & 31, wv = tid >> 5;
  const int half = lane >> 4, lr = lane & 15;
  const int wm = wv >> 1, wn = wv & 1;
  const int m0 = blockIdx.y * BM, n0 = blockIdx.x * BN;
  const unsigned uK = (unsigned)K, uN = (unsigned)N;

  f32x8 zero = {0.f,0.f,0.f,0.f,0.f,0.f,0.f,0.f};
  f32x8 acc[2][4];
#pragma unroll
  for (int mi = 0; mi < 2; mi++)
#pragma unroll
    for (int ni = 0; ni < 4; ni++) acc[mi][ni] = zero;

  for (int kt = 0; kt < K; kt += KT) {
    __syncthreads();
    // ---- stage A tile 128x64 bf16 via async copies (4 x 16B per thread) ----
#pragma unroll
    for (int it = 0; it < 4; it++) {
      int chunk = tid + it * 256;            // 1024 16B chunks
      int r = chunk >> 3, c = (chunk & 7) << 3;
      unsigned off = (unsigned)(m0 + r) * uK + (unsigned)(kt + c);
      async_ld_b128(&As[r][c], bfat(A, off));
      __builtin_prefetch(bfat(A, off + KT), 0, 3);
    }
    // ---- stage B tile into Bs[col][k] ----
    if (Bw) {
      // fp32 weights: 4x4 register-block transpose + convert
      float4 rv[2][4];
#pragma unroll
      for (int v = 0; v < 2; v++) {
        int blk = tid + v * 256;
        int kb = (blk >> 5) << 2;
        int nb = (blk & 31) << 2;
#pragma unroll
        for (int u = 0; u < 4; u++) {
          unsigned off = (unsigned)(kt + kb + u) * uN + (unsigned)(n0 + nb);
          rv[v][u] = *f4at(Bw, off);
          __builtin_prefetch(fatc(Bw, off + (unsigned)KT * uN), 0, 3);
        }
      }
#pragma unroll
      for (int v = 0; v < 2; v++) {
        int blk = tid + v * 256;
        int kb = (blk >> 5) << 2;
        int nb = (blk & 31) << 2;
#pragma unroll
        for (int u = 0; u < 4; u++) {
          bf16x4 col = {(__bf16)f4get(rv[v][0], u), (__bf16)f4get(rv[v][1], u),
                        (__bf16)f4get(rv[v][2], u), (__bf16)f4get(rv[v][3], u)};
          *(bf16x4*)&Bs[nb + u][kb] = col;
        }
      }
    } else {
      // bf16 [N,K]: k-contiguous, async copies
#pragma unroll
      for (int it = 0; it < 4; it++) {
        int chunk = tid + it * 256;
        int r = chunk >> 3, c = (chunk & 7) << 3;
        unsigned off = (unsigned)(n0 + r) * uK + (unsigned)(kt + c);
        async_ld_b128(&Bs[r][c], bfat(Bt, off));
        __builtin_prefetch(bfat(Bt, off + KT), 0, 3);
      }
    }
    wait_async();
    __syncthreads();

    // ---- compute: 2 K-steps of 32, 16 WMMAs ----
#pragma unroll
    for (int ks = 0; ks < 2; ks++) {
      frag_t bfr[4];
#pragma unroll
      for (int ni = 0; ni < 4; ni++) {
        const __bf16* bp = &Bs[wn * 64 + ni * 16 + lr][ks * 32 + half * 8];
        bfr[ni].h[0] = *(const bf16x8*)bp;
        bfr[ni].h[1] = *(const bf16x8*)(bp + 16);
      }
#pragma unroll
      for (int mi = 0; mi < 2; mi++) {
        frag_t af;
        const __bf16* ar = &As[wm * 32 + mi * 16 + lr][ks * 32];
        af.h[0] = *(const bf16x8*)(ar + half * 8);
        af.h[1] = *(const bf16x8*)(ar + 16 + half * 8);
#pragma unroll
        for (int ni = 0; ni < 4; ni++)
          acc[mi][ni] = wmma_bf16(af.v, bfr[ni].v, acc[mi][ni]);
      }
    }
  }

  // ---- epilogue: bias -> act -> residual -> {fp32, bf16} stores ----
#pragma unroll
  for (int mi = 0; mi < 2; mi++)
#pragma unroll
    for (int ni = 0; ni < 4; ni++)
#pragma unroll
      for (int r = 0; r < 8; r++) {
        int row = m0 + wm * 32 + mi * 16 + half * 8 + r;
        int col = n0 + wn * 64 + ni * 16 + lr;
        unsigned co = (unsigned)row * uN + (unsigned)col;
        float v = acc[mi][ni][r];
        if (bias) v += bias[col];
        if (act == 1) v = 0.5f * v * (1.0f + erff(v * 0.70710678f));
        else if (act == 2) v *= 0.125f;
        if (residual) v += *fatc(residual, co);
        if (C) *fat(C, co) = v;
        if (Cbf) *(__bf16*)((char*)Cbf + co * 2u) = (__bf16)v;
      }
}

// ---------------------------------------------------------------------------
__global__ __launch_bounds__(256) void embed_kernel(
    const int* __restrict__ ids, const float* __restrict__ we,
    const float* __restrict__ pe, const float* __restrict__ te,
    float* __restrict__ out, int S)
{
  int t = blockIdx.x;
  int s = t % S;
  int id = ids[t];
  const float* w = we + (size_t)id * D_MODEL;
  const float* p = pe + (size_t)s * D_MODEL;
  float* o = out + (size_t)t * D_MODEL;
  for (int d = threadIdx.x; d < D_MODEL; d += 256) o[d] = w[d] + p[d] + te[d];
}

// LayerNorm; writes fp32 (residual stream) and bf16 shadow (GEMM operand)
__global__ __launch_bounds__(256) void ln_kernel(
    const float* __restrict__ x, const float* __restrict__ g,
    const float* __restrict__ b, float* __restrict__ y, __bf16* __restrict__ yb)
{
  int t = blockIdx.x;
  const float* xr = fatc(x, (unsigned)t * D_MODEL);
  float* yr = fat(y, (unsigned)t * D_MODEL);
  __bf16* ybr = (__bf16*)((char*)yb + (unsigned)t * D_MODEL * 2u);
  int tid = threadIdx.x, lane = tid & 31, wv = tid >> 5;
  float s = 0.f, s2 = 0.f, vals[3];
#pragma unroll
  for (int i = 0; i < 3; i++) {
    float v = xr[tid + i * 256];
    vals[i] = v; s += v; s2 += v * v;
  }
#pragma unroll
  for (int o = 16; o > 0; o >>= 1) { s += __shfl_xor(s, o, 32); s2 += __shfl_xor(s2, o, 32); }
  __shared__ float sr[8], sr2[8];
  if (lane == 0) { sr[wv] = s; sr2[wv] = s2; }
  __syncthreads();
  float tot = 0.f, tot2 = 0.f;
#pragma unroll
  for (int i = 0; i < 8; i++) { tot += sr[i]; tot2 += sr2[i]; }
  float mean = tot * (1.0f / D_MODEL);
  float var = tot2 * (1.0f / D_MODEL) - mean * mean;
  float inv = rsqrtf(var + LN_EPS);
#pragma unroll
  for (int i = 0; i < 3; i++) {
    int d = tid + i * 256;
    float v = (vals[i] - mean) * inv * g[d] + b[d];
    yr[d] = v;
    ybr[d] = (__bf16)v;
  }
}

// ---------------------------------------------------------------------------
// Sliding-window attention (bf16 Q/K/V in, bf16 out), flash-style.
// ---------------------------------------------------------------------------
#define KSP 72

__global__ __launch_bounds__(256) void attn_kernel(
    const __bf16* __restrict__ Qp, const __bf16* __restrict__ Kp,
    const __bf16* __restrict__ Vp, __bf16* __restrict__ Op, int S, int nc)
{
  __shared__ __bf16 Ks[64][KSP];
  __shared__ __bf16 Vt[64][KSP];
  __shared__ __bf16 Ps[8][32][KSP];

  const int hh = blockIdx.x, n = blockIdx.y, b = blockIdx.z;
  const int tid = threadIdx.x, lane = tid & 31, wv = tid >> 5;
  const int half = lane >> 4, lr = lane & 15;
  const int tok0 = b * S + n * CHUNK;

  // Q fragments: direct 16B bf16 loads (A-layout runs are contiguous)
  frag_t qf[2][2];
#pragma unroll
  for (int mi = 0; mi < 2; mi++) {
    unsigned qoff = (unsigned)(tok0 + wv * 32 + mi * 16 + lr) * D_MODEL + hh * DH;
    const __bf16* qr = bfat(Qp, qoff);
#pragma unroll
    for (int ks = 0; ks < 2; ks++) {
      qf[mi][ks].h[0] = *(const bf16x8*)(qr + ks * 32 + half * 8);
      qf[mi][ks].h[1] = *(const bf16x8*)(qr + ks * 32 + 16 + half * 8);
    }
  }

  float rm[2][8], rl[2][8];
  f32x8 zero = {0.f,0.f,0.f,0.f,0.f,0.f,0.f,0.f};
  f32x8 oacc[2][4];
#pragma unroll
  for (int mi = 0; mi < 2; mi++) {
#pragma unroll
    for (int r = 0; r < 8; r++) { rm[mi][r] = -INFINITY; rl[mi][r] = 0.f; }
#pragma unroll
    for (int nt = 0; nt < 4; nt++) oacc[mi][nt] = zero;
  }

  for (int j = 0; j < 3; j++) {
    int nb = n - 1 + j;
    if (nb < 0 || nb >= nc) continue;
    for (int sl = 0; sl < 4; sl++) {
      __syncthreads();
      int ktok = b * S + nb * CHUNK + sl * 64;
      // K: rows contiguous -> async copies (2 x 16B per thread)
#pragma unroll
      for (int it = 0; it < 2; it++) {
        int chunk = tid + it * 256;                 // 512 16B chunks
        int r = chunk >> 3, c = (chunk & 7) << 3;
        unsigned off = (unsigned)(ktok + r) * D_MODEL + (unsigned)(hh * DH + c);
        async_ld_b128(&Ks[r][c], bfat(Kp, off));
        __builtin_prefetch(bfat(Kp, off + 64u * D_MODEL), 0, 3);
      }
      // V: 4x4 bf16 register-block transpose into Vt[dim][key]
      {
        int kb = (tid >> 4) << 2;
        int db = (tid & 15) << 2;
        bf16x4 vr[4];
#pragma unroll
        for (int u = 0; u < 4; u++) {
          unsigned off = (unsigned)(ktok + kb + u) * D_MODEL + (unsigned)(hh * DH + db);
          vr[u] = *(const bf16x4*)bfat(Vp, off);
          __builtin_prefetch(bfat(Vp, off + 64u * D_MODEL), 0, 3);
        }
#pragma unroll
        for (int u = 0; u < 4; u++) {
          bf16x4 col = {vr[0][u], vr[1][u], vr[2][u], vr[3][u]};
          *(bf16x4*)&Vt[db + u][kb] = col;
        }
      }
      wait_async();
      __syncthreads();

      // S = Q x K^T
      f32x8 sfr[2][4];
#pragma unroll
      for (int mi = 0; mi < 2; mi++)
#pragma unroll
        for (int nt = 0; nt < 4; nt++) sfr[mi][nt] = zero;
#pragma unroll
      for (int nt = 0; nt < 4; nt++) {
#pragma unroll
        for (int ks = 0; ks < 2; ks++) {
          frag_t kf;
          const __bf16* kp = &Ks[nt * 16 + lr][ks * 32 + half * 8];
          kf.h[0] = *(const bf16x8*)kp;
          kf.h[1] = *(const bf16x8*)(kp + 16);
          sfr[0][nt] = wmma_bf16(qf[0][ks].v, kf.v, sfr[0][nt]);
          sfr[1][nt] = wmma_bf16(qf[1][ks].v, kf.v, sfr[1][nt]);
        }
      }

      // band mask + online softmax
#pragma unroll
      for (int mi = 0; mi < 2; mi++) {
#pragma unroll
        for (int r = 0; r < 8; r++) {
          int q = wv * 32 + mi * 16 + half * 8 + r;
          float sv[4]; float smax = -INFINITY;
#pragma unroll
          for (int nt = 0; nt < 4; nt++) {
            int kc = sl * 64 + nt * 16 + lr;
            bool valid = (j == 1) || (j == 0 ? (kc >= q) : (kc <= q));
            float sc = valid ? sfr[mi][nt][r] : -INFINITY;
            sv[nt] = sc; smax = fmaxf(smax, sc);
          }
#pragma unroll
          for (int o = 1; o < 16; o <<= 1) smax = fmaxf(smax, __shfl_xor(smax, o, 32));
          float mold = rm[mi][r];
          float mnew = fmaxf(mold, smax);
          float scale = (mnew == -INFINITY) ? 0.f : __expf(mold - mnew);
          float ls = 0.f;
#pragma unroll
          for (int nt = 0; nt < 4; nt++) {
            float p = (sv[nt] == -INFINITY) ? 0.f : __expf(sv[nt] - mnew);
            ls += p;
            Ps[wv][mi * 16 + half * 8 + r][nt * 16 + lr] = (__bf16)p;
          }
#pragma unroll
          for (int o = 1; o < 16; o <<= 1) ls += __shfl_xor(ls, o, 32);
          rl[mi][r] = rl[mi][r] * scale + ls;
          rm[mi][r] = mnew;
#pragma unroll
          for (int nt = 0; nt < 4; nt++) oacc[mi][nt][r] *= scale;
        }
      }

      // O += P x V
      frag_t pf[2][2];
#pragma unroll
      for (int mi = 0; mi < 2; mi++)
#pragma unroll
        for (int ks = 0; ks < 2; ks++) {
          const __bf16* pp = &Ps[wv][mi * 16 + lr][ks * 32];
          pf[mi][ks].h[0] = *(const bf16x8*)(pp + half * 8);
          pf[mi][ks].h[1] = *(const bf16x8*)(pp + 16 + half * 8);
        }
#pragma unroll
      for (int nt = 0; nt < 4; nt++) {
#pragma unroll
        for (int ks = 0; ks < 2; ks++) {
          frag_t vf;
          const __bf16* vp = &Vt[nt * 16 + lr][ks * 32 + half * 8];
          vf.h[0] = *(const bf16x8*)vp;
          vf.h[1] = *(const bf16x8*)(vp + 16);
          oacc[0][nt] = wmma_bf16(pf[0][ks].v, vf.v, oacc[0][nt]);
          oacc[1][nt] = wmma_bf16(pf[1][ks].v, vf.v, oacc[1][nt]);
        }
      }
    }
  }

#pragma unroll
  for (int mi = 0; mi < 2; mi++)
#pragma unroll
    for (int r = 0; r < 8; r++) {
      float inv = rl[mi][r] > 0.f ? 1.0f / rl[mi][r] : 0.f;
      unsigned ooff = (unsigned)(tok0 + wv * 32 + mi * 16 + half * 8 + r) * D_MODEL
                    + (unsigned)(hh * DH);
      __bf16* orow = (__bf16*)((char*)Op + ooff * 2u);
#pragma unroll
      for (int nt = 0; nt < 4; nt++)
        orow[nt * 16 + lr] = (__bf16)(oacc[mi][nt][r] * inv);
    }
}

// ---------------------------------------------------------------------------
__global__ __launch_bounds__(256) void softmax_row_kernel(float* __restrict__ x, int cols)
{
  int row = blockIdx.x;
  float* xr = fat(x, (unsigned)row * (unsigned)cols);
  int tid = threadIdx.x, lane = tid & 31, wv = tid >> 5;
  __shared__ float sr[8];
  float m = -INFINITY;
  for (int c = tid; c < cols; c += 256) m = fmaxf(m, xr[c]);
#pragma unroll
  for (int o = 16; o > 0; o >>= 1) m = fmaxf(m, __shfl_xor(m, o, 32));
  if (lane == 0) sr[wv] = m;
  __syncthreads();
  float rmax = -INFINITY;
#pragma unroll
  for (int i = 0; i < 8; i++) rmax = fmaxf(rmax, sr[i]);
  __syncthreads();
  float s = 0.f;
  for (int c = tid; c < cols; c += 256) { float e = __expf(xr[c] - rmax); xr[c] = e; s += e; }
#pragma unroll
  for (int o = 16; o > 0; o >>= 1) s += __shfl_xor(s, o, 32);
  if (lane == 0) sr[wv] = s;
  __syncthreads();
  float tot = 0.f;
#pragma unroll
  for (int i = 0; i < 8; i++) tot += sr[i];
  float inv = 1.0f / tot;
  for (int c = tid; c < cols; c += 256) xr[c] *= inv;
}

__global__ void colmax_kernel(const float* __restrict__ re, float* __restrict__ out,
                              int rows, int cols)
{
  int b = blockIdx.x;
  int j = threadIdx.x;
  unsigned base = (unsigned)b * (unsigned)rows * (unsigned)cols;
  float m = -INFINITY;
  for (int i = 0; i < rows; i++) m = fmaxf(m, *fatc(re, base + (unsigned)i * cols + j));
  out[(size_t)b * cols + j] = m;
}

__global__ void fc_kernel(const float* __restrict__ rmax, const float* __restrict__ w,
                          const float* __restrict__ bias, float* __restrict__ out)
{
  int t = threadIdx.x;
  if (t < 8) {
    int b = t >> 2, j = t & 3;
    float s = bias[j];
    for (int q = 0; q < 512; q++) s += rmax[b * 512 + q] * w[q * 4 + j];
    out[t] = s;
  }
}

// ---------------------------------------------------------------------------
extern "C" void kernel_launch(void* const* d_in, const int* in_sizes, int n_in,
                              void* d_out, int out_size, void* d_ws, size_t ws_size,
                              hipStream_t stream)
{
  (void)in_sizes; (void)n_in; (void)out_size; (void)ws_size;
  const int* q_ids = (const int*)d_in[0];
  const int* c_ids = (const int*)d_in[1];
  const float* word_emb = (const float*)d_in[4];
  const float* pos_emb  = (const float*)d_in[5];
  const float* type_emb = (const float*)d_in[6];
  const float* eg  = (const float*)d_in[7];
  const float* eb  = (const float*)d_in[8];
  const float* Wq  = (const float*)d_in[9];
  const float* bq  = (const float*)d_in[10];
  const float* Wk  = (const float*)d_in[11];
  const float* bk  = (const float*)d_in[12];
  const float* Wv  = (const float*)d_in[13];
  const float* bv  = (const float*)d_in[14];
  const float* Wo  = (const float*)d_in[15];
  const float* bo  = (const float*)d_in[16];
  const float* g1  = (const float*)d_in[17];
  const float* bln1= (const float*)d_in[18];
  const float* W1  = (const float*)d_in[19];
  const float* bf1 = (const float*)d_in[20];
  const float* W2  = (const float*)d_in[21];
  const float* bf2 = (const float*)d_in[22];
  const float* g2  = (const float*)d_in[23];
  const float* bln2= (const float*)d_in[24];
  const float* fcw = (const float*)d_in[25];
  const float* fcb = (const float*)d_in[26];
  float* outp = (float*)d_out;

  char* wsb = (char*)d_ws;
  size_t off = 0;
  auto alloc = [&](size_t bytes) {
    void* p = wsb + off; off += (bytes + 255) & ~(size_t)255; return p;
  };
  const size_t TQ = 1024, TC = 4096;
  float*  h    = (float*)alloc(TC * D_MODEL * 4);          // fp32 residual stream
  __bf16* hb   = (__bf16*)alloc(TC * D_MODEL * 2);         // bf16 shadow
  __bf16* hqb  = (__bf16*)alloc(TQ * D_MODEL * 2);         // final q bf16
  __bf16* hcb  = (__bf16*)alloc(TC * D_MODEL * 2);         // final c bf16
  __bf16* t0b  = (__bf16*)alloc(TC * D_MODEL * 2);         // q proj
  __bf16* t1b  = (__bf16*)alloc(TC * D_MODEL * 2);         // k proj
  __bf16* t2b  = (__bf16*)alloc(TC * D_MODEL * 2);         // v proj
  __bf16* t3b  = (__bf16*)alloc(TC * D_MODEL * 2);         // attn out
  float*  t4   = (float*)alloc(TC * D_MODEL * 4);          // pre-LN scratch
  __bf16* midb = (__bf16*)alloc(TC * (size_t)FF * 2);      // FFN mid (bf16 only)
  float*  re   = (float*)alloc((size_t)2 * 2048 * 512 * 4);
  float*  rmx  = (float*)alloc((size_t)2 * 512 * 4);

  auto gemm = [&](const __bf16* A, const float* Bw, const __bf16* Bt,
                  const float* bias, const float* res,
                  float* C, __bf16* Cbf, int M, int N, int K, int act) {
    dim3 grid(N / BN, M / BM);
    gemm_kernel<<<grid, dim3(256), 0, stream>>>(A, Bw, Bt, bias, res, C, Cbf, M, N, K, act);
  };

  auto encode = [&](const int* ids, int S, __bf16* final_bf) {
    const int T = 2 * S;
    const int nc = S / CHUNK;
    embed_kernel<<<dim3(T), dim3(256), 0, stream>>>(ids, word_emb, pos_emb, type_emb, t4, S);
    ln_kernel<<<dim3(T), dim3(256), 0, stream>>>(t4, eg, eb, h, hb);
    for (int l = 0; l < NLAYER; l++) {
      const size_t wo = (size_t)l * D_MODEL * D_MODEL;
      gemm(hb, Wq + wo, nullptr, bq + l * D_MODEL, nullptr, nullptr, t0b, T, D_MODEL, D_MODEL, 2);
      gemm(hb, Wk + wo, nullptr, bk + l * D_MODEL, nullptr, nullptr, t1b, T, D_MODEL, D_MODEL, 0);
      gemm(hb, Wv + wo, nullptr, bv + l * D_MODEL, nullptr, nullptr, t2b, T, D_MODEL, D_MODEL, 0);
      attn_kernel<<<dim3(NHEAD, nc, 2), dim3(256), 0, stream>>>(t0b, t1b, t2b, t3b, S, nc);
      gemm(t3b, Wo + wo, nullptr, bo + l * D_MODEL, h, t4, nullptr, T, D_MODEL, D_MODEL, 0);
      ln_kernel<<<dim3(T), dim3(256), 0, stream>>>(t4, g1 + l * D_MODEL, bln1 + l * D_MODEL, h, hb);
      gemm(hb, W1 + (size_t)l * D_MODEL * FF, nullptr, bf1 + l * FF, nullptr,
           nullptr, midb, T, FF, D_MODEL, 1);
      gemm(midb, W2 + (size_t)l * FF * D_MODEL, nullptr, bf2 + l * D_MODEL, h,
           t4, nullptr, T, D_MODEL, FF, 0);
      // last LN also writes the encode's final bf16 buffer
      __bf16* shadow = (l == NLAYER - 1) ? final_bf : hb;
      ln_kernel<<<dim3(T), dim3(256), 0, stream>>>(t4, g2 + l * D_MODEL, bln2 + l * D_MODEL, h, shadow);
      if (l == NLAYER - 1) break;
    }
  };

  encode(q_ids, 512, hqb);
  encode(c_ids, 2048, hcb);

  // re[b] = c_h[b] @ q_h[b]^T  (bf16 A, bf16 transposed B)
  for (int b = 0; b < 2; b++)
    gemm(hcb + (size_t)b * 2048 * D_MODEL, nullptr, hqb + (size_t)b * 512 * D_MODEL,
         nullptr, nullptr, re + (size_t)b * 2048 * 512, nullptr, 2048, 512, 768, 0);

  softmax_row_kernel<<<dim3(2 * 2048), dim3(256), 0, stream>>>(re, 512);
  colmax_kernel<<<dim3(2), dim3(512), 0, stream>>>(re, rmx, 2048, 512);
  fc_kernel<<<dim3(1), dim3(32), 0, stream>>>(rmx, fcw, fcb, outp);
}
